// LongAttention_22445499088906
// MI455X (gfx1250) — compile-verified, matrix operand order checked by reading, not run on hardware
//
#include <hip/hip_runtime.h>
#include <hip/hip_bf16.h>
#include <math.h>

// ---------------------------------------------------------------------------
// Problem constants (from reference)
// ---------------------------------------------------------------------------
#define BB   4
#define TT   4096
#define CC   2048
#define HH   16
#define DH   128
#define KK   4
#define NT   (BB * TT)          // 16384 rows
#define NCHUNK 64               // scan chunks
#define CLEN   (TT / NCHUNK)    // 64 steps per chunk

typedef __attribute__((ext_vector_type(16))) _Float16 v16h;
typedef __attribute__((ext_vector_type(8)))  _Float16 v8h;
typedef __attribute__((ext_vector_type(8)))  float    v8f;

union Frag16 { v16h v; v8h h[2]; };

// gfx1250 async global->LDS path (guarded: falls back to sync loads if the
// toolchain doesn't declare the builtins)
#if defined(__has_builtin)
#if __has_builtin(__builtin_amdgcn_global_load_async_to_lds_b128) && \
    __has_builtin(__builtin_amdgcn_s_wait_asynccnt)
#define USE_ASYNC_LDS 1
#endif
#endif
#ifndef USE_ASYNC_LDS
#define USE_ASYNC_LDS 0
#endif

#if USE_ASYNC_LDS
// Builtin signature (from hipcc diagnostic): param0 is int4* in AS(1) (non-const),
// param1 is the LDS destination, params 2/3 are imm offset and cpol.
typedef int v4i32 __attribute__((vector_size(16)));
__device__ __forceinline__ void async_copy_b128(const _Float16* g, _Float16* l) {
  __builtin_amdgcn_global_load_async_to_lds_b128(
      (__attribute__((address_space(1))) v4i32*)(_Float16*)g,   // drop const, AS1
      (__attribute__((address_space(3))) v4i32*)l,              // addrspacecast to LDS
      0, 0);
}
#endif

// ---------------------------------------------------------------------------
// fp32 -> f16 cast
// ---------------------------------------------------------------------------
__global__ __launch_bounds__(256) void cast_f32_f16(const float* __restrict__ a,
                                                    _Float16* __restrict__ o, int n) {
  int i = blockIdx.x * 256 + threadIdx.x;
  if (i < n) o[i] = (_Float16)a[i];
}

// ---------------------------------------------------------------------------
// causal depthwise conv (K=4, left pad 3) + SiLU; also casts x -> f16
// ---------------------------------------------------------------------------
__global__ __launch_bounds__(256) void conv_silu_cast(const float* __restrict__ x,
                                                      const float* __restrict__ cw,
                                                      const float* __restrict__ cb,
                                                      _Float16* __restrict__ xh,
                                                      _Float16* __restrict__ xch) {
  size_t idx = (size_t)blockIdx.x * 256 + threadIdx.x;
  if (idx >= (size_t)NT * CC) return;
  int c = (int)(idx % CC);
  int t = (int)((idx / CC) % TT);
  size_t rowbase = idx - (size_t)t * CC - c;  // start of (b, t=0)
  float acc = cb[c];
#pragma unroll
  for (int j = 0; j < KK; ++j) {
    int tt = t - (KK - 1) + j;
    if (tt >= 0) acc += x[rowbase + (size_t)tt * CC + c] * cw[c * KK + j];
  }
  float s = acc / (1.0f + __expf(-acc));  // SiLU
  xch[idx] = (_Float16)s;
  xh[idx]  = (_Float16)x[idx];
}

// ---------------------------------------------------------------------------
// WMMA GEMM:  C[N,D] = A[N,K] * W[D,K]^T   (f16 in, f32 out)
// Block tile 128x128, BK=32, 256 threads (8 waves), wave tile 32x64.
// Double-buffered LDS fed by GLOBAL_LOAD_ASYNC_TO_LDS_B128 (ASYNCcnt) when
// available; synchronous b128 global->VGPR->LDS path otherwise.
// ---------------------------------------------------------------------------
#define BM 128
#define BN 128
#define BK 32
#define LDP 48  // padded LDS row (halves); 96B rows keep 16B alignment

__global__ __launch_bounds__(256) void wmma_gemm_nt(const _Float16* __restrict__ A,
                                                    const _Float16* __restrict__ W,
                                                    float* __restrict__ C,
                                                    int N, int K, int D) {
  __shared__ _Float16 As[2][BM][LDP];
  __shared__ _Float16 Bs[2][BN][LDP];

  const int tid  = threadIdx.x;
  const int lane = tid & 31;
  const int wave = tid >> 5;
  const int brow = blockIdx.y * BM;
  const int bcol = blockIdx.x * BN;
  const int wrow = (wave & 3) * 32;   // wave M offset inside block
  const int wcol = (wave >> 2) * 64;  // wave N offset inside block

  // cooperative tile load: 512 x 16B chunks per matrix, 2 per thread
  const int r0 = tid >> 2;            // rows r0 and r0+64, same kq slot
  const int q0 = (tid & 3) * 8;

  v8f acc[2][4];
#pragma unroll
  for (int mi = 0; mi < 2; ++mi)
#pragma unroll
    for (int ni = 0; ni < 4; ++ni) acc[mi][ni] = (v8f){0.f,0.f,0.f,0.f,0.f,0.f,0.f,0.f};

  const int mrow = lane & 15;
  const int kb   = (lane < 16) ? 0 : 8;   // documented 16-bit A/B lane striping

#if USE_ASYNC_LDS
#define ISSUE_TILE(buf, kk)                                                          \
  do {                                                                               \
    async_copy_b128(&A[(size_t)(brow + r0) * K + (kk) + q0],      &As[buf][r0][q0]);      \
    async_copy_b128(&A[(size_t)(brow + r0 + 64) * K + (kk) + q0], &As[buf][r0 + 64][q0]); \
    async_copy_b128(&W[(size_t)(bcol + r0) * K + (kk) + q0],      &Bs[buf][r0][q0]);      \
    async_copy_b128(&W[(size_t)(bcol + r0 + 64) * K + (kk) + q0], &Bs[buf][r0 + 64][q0]); \
  } while (0)

  ISSUE_TILE(0, 0);                       // prologue: tile 0 -> buffer 0
  int cur = 0;
  for (int k0 = 0; k0 < K; k0 += BK) {
    if (k0 + BK < K) {
      ISSUE_TILE(cur ^ 1, k0 + BK);       // stream next tile into other buffer
      __builtin_amdgcn_s_wait_asynccnt(4);  // tile k0's 4 loads done (in-order)
    } else {
      __builtin_amdgcn_s_wait_asynccnt(0);
    }
    __syncthreads();                      // all waves' portions visible

#pragma unroll
    for (int mi = 0; mi < 2; ++mi) {
      Frag16 a;
      a.h[0] = *(const v8h*)&As[cur][wrow + mi * 16 + mrow][kb];
      a.h[1] = *(const v8h*)&As[cur][wrow + mi * 16 + mrow][kb + 16];
#pragma unroll
      for (int ni = 0; ni < 4; ++ni) {
        Frag16 b;
        b.h[0] = *(const v8h*)&Bs[cur][wcol + ni * 16 + mrow][kb];
        b.h[1] = *(const v8h*)&Bs[cur][wcol + ni * 16 + mrow][kb + 16];
        acc[mi][ni] = __builtin_amdgcn_wmma_f32_16x16x32_f16(
            false, a.v, false, b.v, (short)0, acc[mi][ni], false, false);
      }
    }
    __syncthreads();                      // reads done before buffer is reused
    cur ^= 1;
  }
#undef ISSUE_TILE
#else
  // synchronous fallback (single buffer)
  for (int k0 = 0; k0 < K; k0 += BK) {
    *(v8h*)&As[0][r0][q0]      = *(const v8h*)&A[(size_t)(brow + r0) * K + k0 + q0];
    *(v8h*)&As[0][r0 + 64][q0] = *(const v8h*)&A[(size_t)(brow + r0 + 64) * K + k0 + q0];
    *(v8h*)&Bs[0][r0][q0]      = *(const v8h*)&W[(size_t)(bcol + r0) * K + k0 + q0];
    *(v8h*)&Bs[0][r0 + 64][q0] = *(const v8h*)&W[(size_t)(bcol + r0 + 64) * K + k0 + q0];
    if (k0 + BK < K) {
      __builtin_prefetch(&A[(size_t)(brow + r0) * K + k0 + BK], 0, 0);
      __builtin_prefetch(&W[(size_t)(bcol + r0) * K + k0 + BK], 0, 0);
    }
    __syncthreads();
#pragma unroll
    for (int mi = 0; mi < 2; ++mi) {
      Frag16 a;
      a.h[0] = *(const v8h*)&As[0][wrow + mi * 16 + mrow][kb];
      a.h[1] = *(const v8h*)&As[0][wrow + mi * 16 + mrow][kb + 16];
#pragma unroll
      for (int ni = 0; ni < 4; ++ni) {
        Frag16 b;
        b.h[0] = *(const v8h*)&Bs[0][wcol + ni * 16 + mrow][kb];
        b.h[1] = *(const v8h*)&Bs[0][wcol + ni * 16 + mrow][kb + 16];
        acc[mi][ni] = __builtin_amdgcn_wmma_f32_16x16x32_f16(
            false, a.v, false, b.v, (short)0, acc[mi][ni], false, false);
      }
    }
    __syncthreads();
  }
#endif

  // store: lanes 0-15 -> N=lane, M=vgpr; lanes 16-31 -> M=8+vgpr
  const int ncol = lane & 15;
  const int mofs = (lane < 16) ? 0 : 8;
#pragma unroll
  for (int mi = 0; mi < 2; ++mi)
#pragma unroll
    for (int ni = 0; ni < 4; ++ni)
#pragma unroll
      for (int e = 0; e < 8; ++e) {
        size_t row = (size_t)(brow + wrow + mi * 16 + mofs + e);
        size_t col = (size_t)(bcol + wcol + ni * 16 + ncol);
        C[row * D + col] = acc[mi][ni][e];
      }
}

// ---------------------------------------------------------------------------
// per-(b,t,h) l2 norm of q and k; kv = k_norm * v written into v (in place)
// ---------------------------------------------------------------------------
__global__ __launch_bounds__(128) void l2norm_kv(float* __restrict__ q,
                                                 const float* __restrict__ k,
                                                 float* __restrict__ v) {
  const int tid = threadIdx.x;
  const size_t base = (size_t)blockIdx.x * DH;   // blockIdx.x = (b*T+t)*H + h
  float qv = q[base + tid];
  float kv_ = k[base + tid];
  __shared__ float sq[DH], sk[DH];
  sq[tid] = qv * qv;
  sk[tid] = kv_ * kv_;
  __syncthreads();
  for (int s = 64; s > 0; s >>= 1) {
    if (tid < s) { sq[tid] += sq[tid + s]; sk[tid] += sk[tid + s]; }
    __syncthreads();
  }
  float qn = fmaxf(sqrtf(sq[0]), 1e-12f);
  float kn = fmaxf(sqrtf(sk[0]), 1e-12f);
  q[base + tid] = qv / qn;
  v[base + tid] = (kv_ / kn) * v[base + tid];
}

// ---------------------------------------------------------------------------
// chunked decayed scan: S_t = g*S_{t-1} + kv_t
// ---------------------------------------------------------------------------
__global__ __launch_bounds__(256) void scan_pass1(const float* __restrict__ kv,
                                                  const float* __restrict__ gamma,
                                                  float* __restrict__ chunkB) {
  size_t id = (size_t)blockIdx.x * 256 + threadIdx.x;   // [B, NCHUNK, C]
  if (id >= (size_t)BB * NCHUNK * CC) return;
  int hd = (int)(id % CC);
  int rest = (int)(id / CC);
  int chunk = rest % NCHUNK;
  int b = rest / NCHUNK;
  float g = gamma[hd >> 7];
  float acc = 0.f;
  size_t idx = ((size_t)b * TT + (size_t)chunk * CLEN) * CC + hd;
  for (int t = 0; t < CLEN; ++t) { acc = g * acc + kv[idx]; idx += CC; }
  chunkB[id] = acc;
}

__global__ __launch_bounds__(256) void scan_pass2(const float* __restrict__ chunkB,
                                                  const float* __restrict__ gamma,
                                                  float* __restrict__ carry) {
  int id = blockIdx.x * 256 + threadIdx.x;   // [B, C]
  if (id >= BB * CC) return;
  int hd = id % CC;
  int b = id / CC;
  float g = gamma[hd >> 7];
  float gL = powf(g, (float)CLEN);
  float c = 0.f;
  for (int ch = 0; ch < NCHUNK; ++ch) {
    size_t o = ((size_t)b * NCHUNK + ch) * CC + hd;
    carry[o] = c;
    c = gL * c + chunkB[o];
  }
}

__global__ __launch_bounds__(256) void scan_pass3(const float* __restrict__ kv,
                                                  const float* __restrict__ gamma,
                                                  const float* __restrict__ carry,
                                                  const float* __restrict__ gate_raw,
                                                  const float* __restrict__ gate_b,
                                                  const float* __restrict__ q,
                                                  _Float16* __restrict__ outh) {
  size_t id = (size_t)blockIdx.x * 256 + threadIdx.x;   // [B, NCHUNK, C]
  if (id >= (size_t)BB * NCHUNK * CC) return;
  int hd = (int)(id % CC);
  int rest = (int)(id / CC);
  int chunk = rest % NCHUNK;
  int b = rest / NCHUNK;
  float g = gamma[hd >> 7];
  float gb = gate_b[hd];
  float S = carry[id];
  size_t idx = ((size_t)b * TT + (size_t)chunk * CLEN) * CC + hd;
  for (int t = 0; t < CLEN; ++t) {
    S = g * S + kv[idx];
    float gt = 1.0f / (1.0f + __expf(-(gate_raw[idx] + gb)));
    outh[idx] = (_Float16)(gt * S * q[idx]);
    idx += CC;
  }
}

// ---------------------------------------------------------------------------
// LayerNorm over C per row
// ---------------------------------------------------------------------------
__global__ __launch_bounds__(256) void layernorm_rows(const float* __restrict__ y,
                                                      const float* __restrict__ w,
                                                      const float* __restrict__ bias,
                                                      float* __restrict__ out) {
  const int tid = threadIdx.x;
  const size_t base = (size_t)blockIdx.x * CC;
  float r[8];
  float s = 0.f, s2 = 0.f;
#pragma unroll
  for (int i = 0; i < 8; ++i) {
    r[i] = y[base + tid + i * 256];
    s += r[i]; s2 += r[i] * r[i];
  }
  __shared__ float ss[256], sq[256];
  ss[tid] = s; sq[tid] = s2;
  __syncthreads();
  for (int st = 128; st > 0; st >>= 1) {
    if (tid < st) { ss[tid] += ss[tid + st]; sq[tid] += sq[tid + st]; }
    __syncthreads();
  }
  float mu  = ss[0] / (float)CC;
  float var = sq[0] / (float)CC - mu * mu;
  float inv = rsqrtf(var + 1e-5f);
#pragma unroll
  for (int i = 0; i < 8; ++i) {
    int c = tid + i * 256;
    out[base + c] = (r[i] - mu) * inv * w[c] + bias[c];
  }
}

// ---------------------------------------------------------------------------
// launch
// ---------------------------------------------------------------------------
extern "C" void kernel_launch(void* const* d_in, const int* in_sizes, int n_in,
                              void* d_out, int out_size, void* d_ws, size_t ws_size,
                              hipStream_t stream) {
  const float* x      = (const float*)d_in[0];
  const float* Wq     = (const float*)d_in[1];
  const float* Wk     = (const float*)d_in[2];
  const float* Wv     = (const float*)d_in[3];
  const float* Wo     = (const float*)d_in[4];
  const float* conv_w = (const float*)d_in[5];
  const float* conv_b = (const float*)d_in[6];
  const float* gate_w = (const float*)d_in[7];
  const float* gate_b = (const float*)d_in[8];
  const float* ln_w   = (const float*)d_in[9];
  const float* ln_b   = (const float*)d_in[10];
  const float* gamma  = (const float*)d_in[11];
  float* out = (float*)d_out;

  // workspace layout
  char* ws = (char*)d_ws;
  const size_t NTC   = (size_t)NT * CC;
  const size_t szH   = NTC * sizeof(_Float16);      // 64 MB
  const size_t szF   = NTC * sizeof(float);         // 128 MB
  const size_t szWH  = (size_t)CC * CC * sizeof(_Float16); // 8 MB
  const size_t szCh  = (size_t)BB * NCHUNK * CC * sizeof(float); // 2 MB

  _Float16* xh   = (_Float16*)(ws);                 size_t off = szH;
  _Float16* xch  = (_Float16*)(ws + off);           off += szH;
  _Float16* wqh  = (_Float16*)(ws + off);           off += szWH;
  _Float16* wkh  = (_Float16*)(ws + off);           off += szWH;
  _Float16* wvh  = (_Float16*)(ws + off);           off += szWH;
  _Float16* wgh  = (_Float16*)(ws + off);           off += szWH;
  _Float16* woh  = (_Float16*)(ws + off);           off += szWH;
  float* qbuf    = (float*)(ws + off);              off += szF;   // q, later y
  float* kbuf    = (float*)(ws + off);              off += szF;   // k, later gate_raw
  float* vbuf    = (float*)(ws + off);              off += szF;   // v, later kv
  _Float16* outh = (_Float16*)(ws + off);           off += szH;
  float* chunkB  = (float*)(ws + off);              off += szCh;
  float* carry   = (float*)(ws + off);              off += szCh;

  const int nW = CC * CC;
  const int castBlocks = (nW + 255) / 256;
  cast_f32_f16<<<castBlocks, 256, 0, stream>>>(Wq, wqh, nW);
  cast_f32_f16<<<castBlocks, 256, 0, stream>>>(Wk, wkh, nW);
  cast_f32_f16<<<castBlocks, 256, 0, stream>>>(Wv, wvh, nW);
  cast_f32_f16<<<castBlocks, 256, 0, stream>>>(gate_w, wgh, nW);
  cast_f32_f16<<<castBlocks, 256, 0, stream>>>(Wo, woh, nW);

  conv_silu_cast<<<(unsigned)((NTC + 255) / 256), 256, 0, stream>>>(x, conv_w, conv_b, xh, xch);

  dim3 ggrid(CC / BN, NT / BM);   // 16 x 128
  wmma_gemm_nt<<<ggrid, 256, 0, stream>>>(xh, wqh, qbuf, NT, CC, CC);
  wmma_gemm_nt<<<ggrid, 256, 0, stream>>>(xh, wkh, kbuf, NT, CC, CC);
  wmma_gemm_nt<<<ggrid, 256, 0, stream>>>(xh, wvh, vbuf, NT, CC, CC);

  l2norm_kv<<<NT * HH, 128, 0, stream>>>(qbuf, kbuf, vbuf);  // vbuf := k_norm*v

  // gate projection reuses kbuf (k no longer needed)
  wmma_gemm_nt<<<ggrid, 256, 0, stream>>>(xch, wgh, kbuf, NT, CC, CC);

  const int scanN = BB * NCHUNK * CC;
  scan_pass1<<<(scanN + 255) / 256, 256, 0, stream>>>(vbuf, gamma, chunkB);
  scan_pass2<<<(BB * CC + 255) / 256, 256, 0, stream>>>(chunkB, gamma, carry);
  scan_pass3<<<(scanN + 255) / 256, 256, 0, stream>>>(vbuf, gamma, carry, kbuf,
                                                      gate_b, qbuf, outh);

  // output projection reuses qbuf as y
  wmma_gemm_nt<<<ggrid, 256, 0, stream>>>(outh, woh, qbuf, NT, CC, CC);

  layernorm_rows<<<NT, 256, 0, stream>>>(qbuf, ln_w, ln_b, out);
}
